// CNN_69398081568992
// MI455X (gfx1250) — matrix operand order repaired and checked by reading, not compiled, not run on hardware
//
#include <hip/hip_runtime.h>
#include <hip/hip_bf16.h>
#include <stdint.h>

typedef __attribute__((ext_vector_type(16))) _Float16 v16h;
typedef __attribute__((ext_vector_type(8)))  _Float16 v8h;
typedef __attribute__((ext_vector_type(8)))  float    v8f;
typedef __attribute__((ext_vector_type(4)))  float    v4f;

#define BATCH   256
#define HW      19
#define PADHW   21
#define NPIX    (BATCH * HW * HW)      // 92416
#define C_MID   256
#define CP0     32                     // first-layer input channels padded 17->32
#define INV_N   (1.0f / (float)NPIX)
#define BN_EPS  1e-5f

// ---------------------------------------------------------------- utilities
__global__ void zero_words(uint32_t* __restrict__ p, size_t n) {
    size_t t = (size_t)blockIdx.x * blockDim.x + threadIdx.x;
    size_t s = (size_t)gridDim.x * blockDim.x;
    for (; t < n; t += s) p[t] = 0u;
}

__global__ void zero_f32(float* __restrict__ p, int n) {
    int t = blockIdx.x * blockDim.x + threadIdx.x;
    if (t < n) p[t] = 0.0f;
}

// ---------------------------------------------------------------- input build
// padded NHWC f16 [B][21][21][32]; stones -> 1.0 in planes 0..15, color -> plane 16
__global__ void scatter_stones(const int* __restrict__ idx, _Float16* __restrict__ act0) {
    int t = blockIdx.x * blockDim.x + threadIdx.x;          // B*16*128
    if (t >= BATCH * 16 * 128) return;
    int b = t / (16 * 128);
    int rem = t - b * (16 * 128);
    int plane = rem / 128;
    int s = idx[t];
    int y = s / HW, x = s - y * HW;
    act0[((size_t)((b * PADHW + y + 1) * PADHW + x + 1)) * CP0 + plane] = (_Float16)1.0f;
}

__global__ void set_color(const float* __restrict__ color, _Float16* __restrict__ act0) {
    int t = blockIdx.x * blockDim.x + threadIdx.x;          // B*361
    if (t >= NPIX) return;
    int b = t / 361;
    int r = t - b * 361;
    int y = r / HW, x = r - y * HW;
    act0[((size_t)((b * PADHW + y + 1) * PADHW + x + 1)) * CP0 + 16] = (_Float16)color[b];
}

// ---------------------------------------------------------------- weight repack
// W0 [256][17][3][3] f32 -> W0t [9][256][32] f16 (channel-padded with zeros)
__global__ void repack_w0(const float* __restrict__ W0, _Float16* __restrict__ W0t) {
    int t = blockIdx.x * blockDim.x + threadIdx.x;          // 9*256*32
    if (t >= 9 * C_MID * CP0) return;
    int dydx = t / (C_MID * CP0);
    int rem  = t - dydx * (C_MID * CP0);
    int co = rem / CP0;
    int ci = rem - co * CP0;
    float v = (ci < 17) ? W0[(size_t)(co * 17 + ci) * 9 + dydx] : 0.0f;
    W0t[t] = (_Float16)v;
}

// Wm [256][256][3][3] f32 -> Wmt [9][256][256] f16
__global__ void repack_wm(const float* __restrict__ Wm, _Float16* __restrict__ Wmt) {
    int t = blockIdx.x * blockDim.x + threadIdx.x;          // 9*256*256
    if (t >= 9 * C_MID * C_MID) return;
    int dydx = t / (C_MID * C_MID);
    int rem  = t - dydx * (C_MID * C_MID);
    int co = rem / C_MID;
    int ci = rem - co * C_MID;
    Wmt[t] = (_Float16)Wm[(size_t)(co * C_MID + ci) * 9 + dydx];
}

// ---------------------------------------------------------------- implicit-GEMM conv3x3 (WMMA)
// act: padded NHWC f16 [B][21][21][CP]; wt: [9][256][CP] f16; out: NHWC f32 [NPIX][256]
// Wave tile: 32 C_out x 64 pixels (8 accumulators), double-buffered K-loop over
// 9 taps x CP/32 chunks. Fused per-channel sum/sumsq reduction for BatchNorm.
template <int CP>
__global__ __launch_bounds__(256)
void conv3x3_wmma(const _Float16* __restrict__ act,
                  const _Float16* __restrict__ wt,
                  float* __restrict__ out,
                  float* __restrict__ sums /*[512]*/) {
    constexpr int NK  = 9 * (CP / 32);   // K chunks
    constexpr int CPC = CP / 32;

    const int lane = threadIdx.x & 31;
    const int wave = threadIdx.x >> 5;
    const int wid  = blockIdx.x * 8 + wave;
    const int cbase = (wid & 7) * 32;        // 8 C_out tiles of 32
    const int p0    = (wid >> 3) * 64;       // 1444 pixel tiles (shared by all 8 waves of a block)

    const int  n    = lane & 15;
    const bool hi   = lane >= 16;
    const int  aoff = hi ? 8 : 0;            // A: lanes>=16 hold K+8
    const int  boff = 0;                     // B: half offset applied in loadf (K+8 for lanes>=16)

    size_t pixbase[4];
#pragma unroll
    for (int t = 0; t < 4; ++t) {
        int p = p0 + t * 16 + n;
        int b = p / 361;
        int r = p - b * 361;
        int y = r / HW;
        int x = r - y * HW;
        // padded coord of tap (dy,dx) is (y+dy, x+dx) for dy,dx in 0..2
        pixbase[t] = (size_t)((b * PADHW + y) * PADHW + x) * CP + boff;
    }

    v8f acc[8] = {};

    // load fragments for K-chunk k into (A0 rows cbase..+15, A1 rows cbase+16..+31, B[4])
    auto loadf = [&](int k, v16h& A0, v16h& A1, v16h* Bf) {
        int dydx = k / CPC;
        int c0 = (k - dydx * CPC) * 32;
        int dy = dydx / 3, dx = dydx - dy * 3;
        const size_t nb = (size_t)(dy * PADHW + dx) * CP + c0;
        const _Float16* wr = wt + (size_t)dydx * (C_MID * CP)
                                + (size_t)(cbase + n) * CP + c0 + aoff;
        v8h l0 = *(const v8h*)(wr);                     // K = c0+aoff   .. +7
        v8h h0 = *(const v8h*)(wr + 16);                // K = c0+16+aoff.. +7
        v8h l1 = *(const v8h*)(wr + 16 * CP);
        v8h h1 = *(const v8h*)(wr + 16 * CP + 16);
#pragma unroll
        for (int i = 0; i < 8; ++i) {
            A0[i] = l0[i]; A0[i + 8] = h0[i];
            A1[i] = l1[i]; A1[i + 8] = h1[i];
        }
#pragma unroll
        for (int t = 0; t < 4; ++t) {
            v8h bl = *(const v8h*)(act + pixbase[t] + nb + aoff);
            v8h bh = *(const v8h*)(act + pixbase[t] + nb + aoff + 16);
#pragma unroll
            for (int i = 0; i < 8; ++i) { Bf[t][i] = bl[i]; Bf[t][i + 8] = bh[i]; }
        }
    };

    auto domul = [&](const v16h& A0, const v16h& A1, const v16h* Bf) {
#pragma unroll
        for (int t = 0; t < 4; ++t) {
            acc[t]     = __builtin_amdgcn_wmma_f32_16x16x32_f16(
                false, A0, false, Bf[t], (short)0, acc[t],     false, false);
            acc[4 + t] = __builtin_amdgcn_wmma_f32_16x16x32_f16(
                false, A1, false, Bf[t], (short)0, acc[4 + t], false, false);
        }
    };

    // software-pipelined (double-buffered) K loop
    v16h A0a, A1a, Ba[4], A0b, A1b, Bb[4];
    loadf(0, A0a, A1a, Ba);
    for (int k = 0; k + 2 <= NK; k += 2) {
        loadf(k + 1, A0b, A1b, Bb);
        domul(A0a, A1a, Ba);
        if (k + 2 < NK) loadf(k + 2, A0a, A1a, Ba);
        domul(A0b, A1b, Bb);
    }
    if (NK & 1) domul(A0a, A1a, Ba);   // NK=9 tail (last chunk sits in buffer a)

    // ---- store conv output (8 consecutive channels per lane per accumulator)
#pragma unroll
    for (int cg = 0; cg < 2; ++cg) {
#pragma unroll
        for (int t = 0; t < 4; ++t) {
            int p = p0 + t * 16 + n;
            float* o = out + (size_t)p * C_MID + cbase + cg * 16 + aoff;
            v8f a = acc[cg * 4 + t];
            v4f lo = { a[0], a[1], a[2], a[3] };
            v4f h4 = { a[4], a[5], a[6], a[7] };
            *(v4f*)o = lo;
            *(v4f*)(o + 4) = h4;
        }
    }

    // ---- fused BN reduction: per-channel sum/sumsq over this wave's 64 pixels
    float ps[16], pq[16];
#pragma unroll
    for (int cg = 0; cg < 2; ++cg) {
#pragma unroll
        for (int r = 0; r < 8; ++r) {
            float s = 0.0f, q = 0.0f;
#pragma unroll
            for (int t = 0; t < 4; ++t) {
                float v = acc[cg * 4 + t][r];
                s += v; q += v * v;
            }
            ps[cg * 8 + r] = s;
            pq[cg * 8 + r] = q;
        }
    }
    // reduce across the 16 lanes of each half (pixels); halves hold distinct channels
#pragma unroll
    for (int m = 1; m <= 8; m <<= 1) {
#pragma unroll
        for (int i = 0; i < 16; ++i) {
            ps[i] += __shfl_xor(ps[i], m, 32);
            pq[i] += __shfl_xor(pq[i], m, 32);
        }
    }
    if (n == 0) {   // lanes 0 and 16 write their half's channels
#pragma unroll
        for (int cg = 0; cg < 2; ++cg) {
#pragma unroll
            for (int r = 0; r < 8; ++r) {
                int c = cbase + cg * 16 + aoff + r;
                atomicAdd(&sums[c],         ps[cg * 8 + r]);
                atomicAdd(&sums[C_MID + c], pq[cg * 8 + r]);
            }
        }
    }
}

// ---------------------------------------------------------------- BatchNorm pipeline
__global__ void bn_stats(const float* __restrict__ sums,
                         const float* __restrict__ g, const float* __restrict__ be,
                         float* __restrict__ ss /*[512] scale,shift*/) {
    int c = threadIdx.x;                  // 256 threads
    float m  = sums[c] * INV_N;
    float v  = sums[C_MID + c] * INV_N - m * m;
    float sc = g[c] * rsqrtf(v + BN_EPS);
    ss[c]         = sc;
    ss[C_MID + c] = be[c] - m * sc;
}

// BN + (optional residual) + ReLU, repack f32 NHWC -> padded f16 NHWC
__global__ __launch_bounds__(256)
void bn_apply(const float* __restrict__ y, const float* __restrict__ ss,
              const _Float16* __restrict__ resid, int has_resid,
              _Float16* __restrict__ dst) {
    int c = threadIdx.x;
    float sc = ss[c], sh = ss[C_MID + c];
    for (int p = blockIdx.x; p < NPIX; p += gridDim.x) {
        int b = p / 361;
        int r = p - b * 361;
        int yy = r / HW + 1, xx = r - (yy - 1) * HW + 1;
        size_t di = (size_t)((b * PADHW + yy) * PADHW + xx) * C_MID + c;
        float v = y[(size_t)p * C_MID + c] * sc + sh;
        if (has_resid) v += (float)resid[di];
        v = v > 0.0f ? v : 0.0f;
        dst[di] = (_Float16)v;
    }
}

// ---------------------------------------------------------------- 1x1 head conv + BN stats
__global__ __launch_bounds__(256)
void head_conv(const _Float16* __restrict__ R, const float* __restrict__ We,
               float* __restrict__ out1, float* __restrict__ s2 /*[4]*/) {
    int lane = threadIdx.x & 31;
    int wave = threadIdx.x >> 5;
    int p = blockIdx.x * 8 + wave;
    if (p >= NPIX) return;
    int b = p / 361;
    int r = p - b * 361;
    int yy = r / HW + 1, xx = r - (yy - 1) * HW + 1;
    const _Float16* row = R + (size_t)((b * PADHW + yy) * PADHW + xx) * C_MID;
    float s = 0.0f;
    for (int c = lane; c < C_MID; c += 32) s += (float)row[c] * We[c];
#pragma unroll
    for (int off = 16; off > 0; off >>= 1) s += __shfl_xor(s, off, 32);
    if (lane == 0) {
        out1[p] = s;
        atomicAdd(&s2[0], s);
        atomicAdd(&s2[1], s * s);
    }
}

__global__ void head_stats(float* __restrict__ s2,
                           const float* __restrict__ ge, const float* __restrict__ bee) {
    float m  = s2[0] * INV_N;
    float v  = s2[1] * INV_N - m * m;
    float sc = ge[0] * rsqrtf(v + BN_EPS);
    s2[2] = sc;
    s2[3] = bee[0] - m * sc;
}

__global__ void head_apply(const float* __restrict__ out1, const float* __restrict__ s2,
                           float* __restrict__ out) {
    int p = blockIdx.x * blockDim.x + threadIdx.x;
    if (p >= NPIX) return;
    float v = out1[p] * s2[2] + s2[3];
    out[p] = v > 0.0f ? v : 0.0f;
}

// ---------------------------------------------------------------- host launch
extern "C" void kernel_launch(void* const* d_in, const int* in_sizes, int n_in,
                              void* d_out, int out_size, void* d_ws, size_t ws_size,
                              hipStream_t stream) {
    (void)in_sizes; (void)n_in; (void)out_size; (void)ws_size;

    const int*   stone = (const int*)  d_in[0];
    const float* color = (const float*)d_in[1];
    const float* W0    = (const float*)d_in[2];
    // d_in[3] = b0, d_in[7] = bm, d_in[11] = bE  (conv biases cancel exactly under BN)
    const float* g0    = (const float*)d_in[4];
    const float* be0   = (const float*)d_in[5];
    const float* Wm    = (const float*)d_in[6];
    const float* gm    = (const float*)d_in[8];
    const float* bem   = (const float*)d_in[9];
    const float* We    = (const float*)d_in[10];
    const float* ge    = (const float*)d_in[12];
    const float* bee   = (const float*)d_in[13];
    float* out = (float*)d_out;

    // ---- workspace layout (bytes)
    char* ws = (char*)d_ws;
    const size_t SZ_ACT0 = (size_t)BATCH * PADHW * PADHW * CP0   * 2;  //  7,225,344
    const size_t SZ_ACT  = (size_t)BATCH * PADHW * PADHW * C_MID * 2;  // 57,802,752
    const size_t SZ_CONV = (size_t)NPIX * C_MID * 4;                   // 94,633,984
    _Float16* act0 = (_Float16*)(ws);
    _Float16* actR = (_Float16*)(ws + SZ_ACT0);
    _Float16* actD = (_Float16*)(ws + SZ_ACT0 + SZ_ACT);
    float*    conv = (float*)   (ws + SZ_ACT0 + 2 * SZ_ACT);
    _Float16* w0t  = (_Float16*)(ws + SZ_ACT0 + 2 * SZ_ACT + SZ_CONV);
    _Float16* wmt  = (_Float16*)((char*)w0t + (size_t)9 * C_MID * CP0 * 2);
    float*    sums = (float*)   ((char*)wmt + (size_t)9 * C_MID * C_MID * 2);   // [512]
    float*    ss   = sums + 512;                                                // [512]
    float*    out1 = ss + 512;                                                  // [NPIX]
    float*    s2   = out1 + NPIX;                                               // [4]

    // ---- zero padded activation buffers (contiguous region)
    {
        size_t nwords = (SZ_ACT0 + 2 * SZ_ACT) / 4;
        zero_words<<<2048, 256, 0, stream>>>((uint32_t*)ws, nwords);
    }

    // ---- weight repack + input build
    repack_w0<<<(9 * C_MID * CP0 + 255) / 256, 256, 0, stream>>>(W0, w0t);
    repack_wm<<<(9 * C_MID * C_MID + 255) / 256, 256, 0, stream>>>(Wm, wmt);
    scatter_stones<<<(BATCH * 16 * 128 + 255) / 256, 256, 0, stream>>>(stone, act0);
    set_color<<<(NPIX + 255) / 256, 256, 0, stream>>>(color, act0);

    const int GEMM_BLOCKS = (8 * (NPIX / 64)) / 8;   // 1444

    // ---- layer 0: conv(17->256) + BN + ReLU -> residual R
    zero_f32<<<2, 256, 0, stream>>>(sums, 512);
    conv3x3_wmma<CP0><<<GEMM_BLOCKS, 256, 0, stream>>>(act0, w0t, conv, sums);
    bn_stats<<<1, 256, 0, stream>>>(sums, g0, be0, ss);
    bn_apply<<<2048, 256, 0, stream>>>(conv, ss, nullptr, 0, actR);

    // ---- 3 residual blocks (shared mid weights)
    for (int blk = 0; blk < 3; ++blk) {
        zero_f32<<<2, 256, 0, stream>>>(sums, 512);
        conv3x3_wmma<C_MID><<<GEMM_BLOCKS, 256, 0, stream>>>(actR, wmt, conv, sums);
        bn_stats<<<1, 256, 0, stream>>>(sums, gm, bem, ss);
        bn_apply<<<2048, 256, 0, stream>>>(conv, ss, nullptr, 0, actD);

        zero_f32<<<2, 256, 0, stream>>>(sums, 512);
        conv3x3_wmma<C_MID><<<GEMM_BLOCKS, 256, 0, stream>>>(actD, wmt, conv, sums);
        bn_stats<<<1, 256, 0, stream>>>(sums, gm, bem, ss);
        bn_apply<<<2048, 256, 0, stream>>>(conv, ss, actR, 1, actR);  // in-place residual update
    }

    // ---- head: 1x1 conv + BN + ReLU
    zero_f32<<<1, 256, 0, stream>>>(s2, 4);
    head_conv<<<(NPIX + 7) / 8, 256, 0, stream>>>(actR, We, out1, s2);
    head_stats<<<1, 1, 0, stream>>>(s2, ge, bee);
    head_apply<<<(NPIX + 255) / 256, 256, 0, stream>>>(out1, s2, out);
}
